// LocalContextProcessor_45698452030312
// MI455X (gfx1250) — matrix operand order repaired and checked by reading, not compiled, not run on hardware
//
#include <hip/hip_runtime.h>
#include <cstdint>
#include <cstddef>

// ---------------------------------------------------------------------------
// CDNA5 (gfx1250) local linear attention, bf16 WMMA pipeline, double-buffered,
// async global->LDS staging where the copy needs no conversion.
//   K1: QKV = X @ W_qkv  (elu+1 on q,k cols)        -> ws (bf16, 16384x3072)
//   K2: per (window,head): ctx = k^T v ; out = q ctx -> overwrite q region
//   K3: Y = attn @ W_out + b_out (fp32 out)
// ---------------------------------------------------------------------------

typedef __attribute__((ext_vector_type(8)))  unsigned int v8u;
typedef __attribute__((ext_vector_type(16))) __bf16       v16bf;
typedef __attribute__((ext_vector_type(8)))  float        v8f;
typedef __attribute__((ext_vector_type(4)))  int          v4i;

typedef __attribute__((address_space(1))) v4i* gv4i_ptr;   // global int4*
typedef __attribute__((address_space(3))) v4i* lv4i_ptr;   // LDS int4*

#define N_TOK   16384
#define DIM     1024
#define N3      3072
#define WS_     256
#define HEADS   8
#define DHEAD   128

#if __has_builtin(__builtin_amdgcn_global_load_async_to_lds_b128)
#define HAVE_ASYNC_LDS 1
#else
#define HAVE_ASYNC_LDS 0
#endif

// Pure-copy staging: async DMA into LDS when available (ASYNCcnt-tracked),
// else a synchronous 128-bit reg round-trip.
__device__ __forceinline__ void async_copy_b128(unsigned short* ldsDst,
                                                const unsigned short* gSrc) {
#if HAVE_ASYNC_LDS
    __builtin_amdgcn_global_load_async_to_lds_b128(
        (gv4i_ptr)gSrc, (lv4i_ptr)ldsDst, 0, 0);
#else
    *(uint4*)ldsDst = *(const uint4*)gSrc;
#endif
}
__device__ __forceinline__ void wait_async() {
#if HAVE_ASYNC_LDS
#if __has_builtin(__builtin_amdgcn_s_wait_asynccnt)
    __builtin_amdgcn_s_wait_asynccnt(0);
#else
    asm volatile("s_wait_asynccnt 0x0" ::: "memory");
#endif
#endif
}

// Native f32 -> bf16 (fptrunc, RNE) -> v_cvt_pk_bf16_f32
__device__ __forceinline__ unsigned short f2bf(float f) {
    return __builtin_bit_cast(unsigned short, (__bf16)f);
}
__device__ __forceinline__ unsigned pack2bf(float lo, float hi) {
    return (unsigned)f2bf(lo) | ((unsigned)f2bf(hi) << 16);
}

__device__ __forceinline__ v8f wmma_bf16(v8u a, v8u b, v8f c) {
    return __builtin_amdgcn_wmma_f32_16x16x32_bf16(
        false, __builtin_bit_cast(v16bf, a),
        false, __builtin_bit_cast(v16bf, b),
        (short)0, c, false, false);
}
__device__ __forceinline__ v8f vzero() {
    v8f z;
#pragma unroll
    for (int i = 0; i < 8; ++i) z[i] = 0.0f;
    return z;
}

// A-fragment (16x32 bf16, K contiguous in LDS row): ISA 7.12.2 pairing.
__device__ __forceinline__ v8u load_fragA(const unsigned* lds32, int row,
                                          int stride32, int kOff32, int laneHi) {
    v8u u;
#pragma unroll
    for (int i = 0; i < 8; ++i)
        u[i] = lds32[row * stride32 + kOff32 + (i < 4 ? i : i + 4) + laneHi * 4];
    return u;
}
// B-fragment (32x16 bf16) from K-transposed LDS ([N][K], K contiguous).
__device__ __forceinline__ v8u load_fragB(const unsigned* lds32, int col,
                                          int stride32, int kOff32, int laneHi) {
    v8u u;
#pragma unroll
    for (int r = 0; r < 8; ++r)
        u[r] = lds32[col * stride32 + kOff32 + (r < 4 ? r : r + 4) + laneHi * 4];
    return u;
}

// ---------------------------------------------------------------------------
// Kernel 1: QKV GEMM.  grid (24, 128), block 256 (8 waves, 2x4 wave grid)
// (both tiles need fp32->bf16 conversion, so staging stays through VGPRs)
// ---------------------------------------------------------------------------
__global__ __launch_bounds__(256)
void qkv_gemm_kernel(const float* __restrict__ X, const float* __restrict__ Wqkv,
                     unsigned short* __restrict__ QKV) {
    const int t = threadIdx.x, lane = t & 31, wid = t >> 5;
    const int wm = wid >> 2, wn = wid & 3, laneHi = lane >> 4, laneLo = lane & 15;
    const int n0 = blockIdx.x * 128;
    const int m0 = blockIdx.y * 128;

    __shared__ unsigned short lds_a[2][128 * 32];   // [m][k] bf16, 2x8 KB
    __shared__ unsigned short lds_b[2][128 * 32];   // [n][k] bf16 transposed

    const float* px = X + (size_t)(m0 + (t >> 3)) * DIM + ((t & 7) << 2);
    const float* pw = Wqkv + (size_t)(t >> 5) * N3 + n0 + ((t & 31) << 2);
    const int laOff = (t >> 3) * 16 + ((t & 7) << 1);        // u32 units
    const int lbOff = ((t & 31) << 2) * 32 + (t >> 5);       // u16 units

    float4 gx[4], gw[4];
    auto fetch = [&]() {
#pragma unroll
        for (int j = 0; j < 4; ++j) gx[j] = *(const float4*)(px + (size_t)j * 32 * DIM);
#pragma unroll
        for (int j = 0; j < 4; ++j) gw[j] = *(const float4*)(pw + (size_t)j * 8 * N3);
        px += 32;
        pw += (size_t)32 * N3;
    };
    auto stage = [&](int buf) {
        unsigned* la32 = (unsigned*)lds_a[buf] + laOff;
        unsigned short* lb = lds_b[buf] + lbOff;
#pragma unroll
        for (int j = 0; j < 4; ++j) {
            la32[j * 512]     = pack2bf(gx[j].x, gx[j].y);
            la32[j * 512 + 1] = pack2bf(gx[j].z, gx[j].w);
        }
#pragma unroll
        for (int j = 0; j < 4; ++j) {
            lb[j * 8 + 0 * 32] = f2bf(gw[j].x);
            lb[j * 8 + 1 * 32] = f2bf(gw[j].y);
            lb[j * 8 + 2 * 32] = f2bf(gw[j].z);
            lb[j * 8 + 3 * 32] = f2bf(gw[j].w);
        }
    };

    v8f acc[4][2];
#pragma unroll
    for (int a = 0; a < 4; ++a)
#pragma unroll
        for (int b = 0; b < 2; ++b) acc[a][b] = vzero();

    fetch();
    stage(0);
    const int NK = DIM / 32;
    for (int kt = 0; kt < NK; ++kt) {
        const int cur = kt & 1;
        __syncthreads();
        if (kt + 1 < NK) fetch();
        const unsigned* la32 = (const unsigned*)lds_a[cur];
        const unsigned* lb32 = (const unsigned*)lds_b[cur];
        v8u afr[4], bfr[2];
#pragma unroll
        for (int a = 0; a < 4; ++a)
            afr[a] = load_fragA(la32, wm * 64 + a * 16 + laneLo, 16, 0, laneHi);
#pragma unroll
        for (int b = 0; b < 2; ++b)
            bfr[b] = load_fragB(lb32, wn * 32 + b * 16 + laneLo, 16, 0, laneHi);
#pragma unroll
        for (int a = 0; a < 4; ++a)
#pragma unroll
            for (int b = 0; b < 2; ++b)
                acc[a][b] = wmma_bf16(afr[a], bfr[b], acc[a][b]);
        if (kt + 1 < NK) stage(cur ^ 1);
    }

    // Epilogue: branchless elu(x)+1 on q,k segments (cols < 2048); bf16 store.
#pragma unroll
    for (int b = 0; b < 2; ++b) {
        const int col = n0 + wn * 32 + b * 16 + laneLo;
        const bool act = col < 2048;
#pragma unroll
        for (int a = 0; a < 4; ++a) {
            unsigned short* p =
                QKV + (size_t)(m0 + wm * 64 + a * 16 + laneHi * 8) * N3 + col;
#pragma unroll
            for (int r = 0; r < 8; ++r) {
                float x  = acc[a][b][r];
                float xe = __expf(fminf(x, 0.0f));
                float xa = (x > 0.0f) ? (x + 1.0f) : xe;
                x = act ? xa : x;
                p[(size_t)r * N3] = f2bf(x);   // r*6144B folds into inst offset
            }
        }
    }
}

// ---------------------------------------------------------------------------
// Kernel 2: per-(window,head) linear attention.  grid (8, 64), block 256.
// ---------------------------------------------------------------------------
__global__ __launch_bounds__(256)
void attn_kernel(unsigned short* __restrict__ QKV) {
    const int t = threadIdx.x, lane = t & 31, wid = t >> 5;
    const int laneHi = lane >> 4, laneLo = lane & 15;
    const int h = blockIdx.x;                   // 0..7
    const int w = blockIdx.y;                   // 0..63
    const size_t rowBase = (size_t)w * WS_ * N3;
    const int headOff = h * DHEAD;

    __shared__ unsigned short lds_ctx[128 * 128];    // ctx bf16 [e][d], 32 KB
    __shared__ unsigned short lds_buf[2][64 * 128];  // 2x16 KB (kT|vT, then q)
    unsigned* ctx32 = (unsigned*)lds_ctx;

    const int wm = wid >> 2, wn = wid & 3;

    // ---- Stage 1: ctx[d][e] = sum_n k[n][d] * v[n][e]  (M=128,N=128,K=256)
    // transposed scatter -> must pass through VGPRs (no async path)
    const unsigned short* pk =
        QKV + rowBase + (size_t)(t >> 4) * N3 + headOff + ((t & 15) << 3) + 1024;
    const unsigned short* pv = pk + 1024;
    const int kvOff = ((t & 15) << 3) * 32 + (t >> 4);       // d*32 + n, u16 units

    uint4 gk4[2], gv4[2];
    auto fetch_kv = [&]() {
#pragma unroll
        for (int j = 0; j < 2; ++j) {
            gk4[j] = *(const uint4*)(pk + (size_t)j * 16 * N3);
            gv4[j] = *(const uint4*)(pv + (size_t)j * 16 * N3);
        }
        pk += (size_t)32 * N3;
        pv += (size_t)32 * N3;
    };
    auto scatter8 = [](unsigned short* base, uint4 v) {
        base[0 * 32] = (unsigned short)(v.x);
        base[1 * 32] = (unsigned short)(v.x >> 16);
        base[2 * 32] = (unsigned short)(v.y);
        base[3 * 32] = (unsigned short)(v.y >> 16);
        base[4 * 32] = (unsigned short)(v.z);
        base[5 * 32] = (unsigned short)(v.z >> 16);
        base[6 * 32] = (unsigned short)(v.w);
        base[7 * 32] = (unsigned short)(v.w >> 16);
    };
    auto stage_kv = [&](int buf) {
        unsigned short* lb = lds_buf[buf] + kvOff;
#pragma unroll
        for (int j = 0; j < 2; ++j) {
            scatter8(lb + j * 16, gk4[j]);          // kT[d][n]
            scatter8(lb + 4096 + j * 16, gv4[j]);   // vT[e][n]
        }
    };

    v8f accc[4][2];
#pragma unroll
    for (int a = 0; a < 4; ++a)
#pragma unroll
        for (int b = 0; b < 2; ++b) accc[a][b] = vzero();

    fetch_kv();
    stage_kv(0);
    for (int it = 0; it < 8; ++it) {
        const int cur = it & 1;
        __syncthreads();
        if (it + 1 < 8) fetch_kv();
        const unsigned* la32 = (const unsigned*)lds_buf[cur];
        v8u afr[4], bfr[2];
#pragma unroll
        for (int a = 0; a < 4; ++a)
            afr[a] = load_fragA(la32, wm * 64 + a * 16 + laneLo, 16, 0, laneHi);
#pragma unroll
        for (int b = 0; b < 2; ++b)
            bfr[b] = load_fragB(la32 + 2048, wn * 32 + b * 16 + laneLo, 16, 0, laneHi);
#pragma unroll
        for (int a = 0; a < 4; ++a)
#pragma unroll
            for (int b = 0; b < 2; ++b)
                accc[a][b] = wmma_bf16(afr[a], bfr[b], accc[a][b]);
        if (it + 1 < 8) stage_kv(cur ^ 1);
    }

    // ctx -> LDS bf16, transposed [e][d]; d consecutive -> one b128 per frag
#pragma unroll
    for (int a = 0; a < 4; ++a)
#pragma unroll
        for (int b = 0; b < 2; ++b) {
            int d0 = wm * 64 + a * 16 + laneHi * 8;
            int e  = wn * 32 + b * 16 + laneLo;
            uint4 pkv;
            pkv.x = pack2bf(accc[a][b][0], accc[a][b][1]);
            pkv.y = pack2bf(accc[a][b][2], accc[a][b][3]);
            pkv.z = pack2bf(accc[a][b][4], accc[a][b][5]);
            pkv.w = pack2bf(accc[a][b][6], accc[a][b][7]);
            *(uint4*)&lds_ctx[e * 128 + d0] = pkv;
        }
    __syncthreads();

    // ---- Stage 2: out[n][e] = sum_d q[n][d] * ctx[d][e]  (K=128)
    const int e0 = wid * 16;
    v8u cfr[8];
#pragma unroll
    for (int ks = 0; ks < 8; ++ks)
        cfr[ks] = load_fragB(ctx32, e0 + laneLo, 64, ks * 16, laneHi);

    // q staging is a pure bf16 copy -> async global->LDS (ASYNCcnt path).
    const unsigned short* pq =
        QKV + rowBase + (size_t)(t >> 4) * N3 + headOff + ((t & 15) << 3);
    const int qOff = (t >> 4) * 128 + ((t & 15) << 3);       // n*128 + d, u16 units

    auto issue_q = [&](int buf) {
        unsigned short* lb = lds_buf[buf] + qOff;
#pragma unroll
        for (int j = 0; j < 4; ++j)
            async_copy_b128(lb + j * 16 * 128, pq + (size_t)j * 16 * N3);
        pq += (size_t)64 * N3;
    };

    issue_q(0);
    for (int c = 0; c < 4; ++c) {
        const int cur = c & 1;
        wait_async();                 // my async fills of buf[cur] landed
        __syncthreads();              // everyone's landed / old readers done
        if (c + 1 < 4) issue_q(cur ^ 1);   // overlaps compute below
        const unsigned* lq32 = (const unsigned*)lds_buf[cur];
        v8f acco[4];
#pragma unroll
        for (int a = 0; a < 4; ++a) acco[a] = vzero();
#pragma unroll
        for (int ks = 0; ks < 8; ++ks)
#pragma unroll
            for (int a = 0; a < 4; ++a) {
                v8u afr = load_fragA(lq32, a * 16 + laneLo, 64, ks * 16, laneHi);
                acco[a] = wmma_bf16(afr, cfr[ks], acco[a]);
            }
        // write attention output in place over the (already consumed) q chunk
#pragma unroll
        for (int a = 0; a < 4; ++a) {
            unsigned short* p = QKV + rowBase +
                (size_t)(c * 64 + a * 16 + laneHi * 8) * N3 + headOff + e0 + laneLo;
#pragma unroll
            for (int r = 0; r < 8; ++r)
                p[(size_t)r * N3] = f2bf(acco[a][r]);
        }
    }
}

// ---------------------------------------------------------------------------
// Kernel 3: Y = attn @ W_out + b_out (fp32 out).  grid (8, 128), block 256.
// A tile is a pure bf16 copy -> async global->LDS; W tile needs cvt -> VGPRs.
// ---------------------------------------------------------------------------
__global__ __launch_bounds__(256)
void out_gemm_kernel(const unsigned short* __restrict__ ATT, const float* __restrict__ Wout,
                     const float* __restrict__ bias, float* __restrict__ OUT) {
    const int t = threadIdx.x, lane = t & 31, wid = t >> 5;
    const int wm = wid >> 2, wn = wid & 3, laneHi = lane >> 4, laneLo = lane & 15;
    const int n0 = blockIdx.x * 128;
    const int m0 = blockIdx.y * 128;

    __shared__ unsigned short lds_a[2][128 * 32];
    __shared__ unsigned short lds_b[2][128 * 32];

    const unsigned short* pa = ATT + (size_t)(m0 + (t >> 2)) * N3 + ((t & 3) << 3);
    const float* pw = Wout + (size_t)(t >> 5) * DIM + n0 + ((t & 31) << 2);
    const int laOff = (t >> 2) * 32 + ((t & 3) << 3);        // u16 units
    const int lbOff = ((t & 31) << 2) * 32 + (t >> 5);       // u16 units

    auto issueA = [&](int buf) {
        unsigned short* la = lds_a[buf] + laOff;
#pragma unroll
        for (int j = 0; j < 2; ++j)
            async_copy_b128(la + j * 64 * 32, pa + (size_t)j * 64 * N3);
        pa += 32;
    };

    float4 gw[4];
    auto fetchW = [&]() {
#pragma unroll
        for (int j = 0; j < 4; ++j)
            gw[j] = *(const float4*)(pw + (size_t)j * 8 * DIM);
        pw += (size_t)32 * DIM;
    };
    auto stageW = [&](int buf) {
        unsigned short* lb = lds_b[buf] + lbOff;
#pragma unroll
        for (int j = 0; j < 4; ++j) {
            lb[j * 8 + 0 * 32] = f2bf(gw[j].x);
            lb[j * 8 + 1 * 32] = f2bf(gw[j].y);
            lb[j * 8 + 2 * 32] = f2bf(gw[j].z);
            lb[j * 8 + 3 * 32] = f2bf(gw[j].w);
        }
    };

    v8f acc[4][2];
#pragma unroll
    for (int a = 0; a < 4; ++a)
#pragma unroll
        for (int b = 0; b < 2; ++b) acc[a][b] = vzero();

    issueA(0);
    fetchW();
    stageW(0);
    const int NK = DIM / 32;
    for (int kt = 0; kt < NK; ++kt) {
        const int cur = kt & 1;
        wait_async();                 // async fills of lds_a[cur] landed
        __syncthreads();
        if (kt + 1 < NK) {
            issueA(cur ^ 1);          // async next A tile (overlaps compute)
            fetchW();                 // W tile to regs (needs cvt)
        }
        const unsigned* la32 = (const unsigned*)lds_a[cur];
        const unsigned* lb32 = (const unsigned*)lds_b[cur];
        v8u afr[4], bfr[2];
#pragma unroll
        for (int a = 0; a < 4; ++a)
            afr[a] = load_fragA(la32, wm * 64 + a * 16 + laneLo, 16, 0, laneHi);
#pragma unroll
        for (int b = 0; b < 2; ++b)
            bfr[b] = load_fragB(lb32, wn * 32 + b * 16 + laneLo, 16, 0, laneHi);
#pragma unroll
        for (int a = 0; a < 4; ++a)
#pragma unroll
            for (int b = 0; b < 2; ++b)
                acc[a][b] = wmma_bf16(afr[a], bfr[b], acc[a][b]);
        if (kt + 1 < NK) stageW(cur ^ 1);
    }

#pragma unroll
    for (int b = 0; b < 2; ++b) {
        const int col = n0 + wn * 32 + b * 16 + laneLo;
        const float bv = bias[col];
#pragma unroll
        for (int a = 0; a < 4; ++a) {
            float* p = OUT + (size_t)(m0 + wm * 64 + a * 16 + laneHi * 8) * DIM + col;
#pragma unroll
            for (int r = 0; r < 8; ++r)
                p[(size_t)r * DIM] = acc[a][b][r] + bv;    // r*4096B inst offset
        }
    }
}

// ---------------------------------------------------------------------------
extern "C" void kernel_launch(void* const* d_in, const int* in_sizes, int n_in,
                              void* d_out, int out_size, void* d_ws, size_t ws_size,
                              hipStream_t stream) {
    const float* x    = (const float*)d_in[0];   // (4,4096,1024)
    const float* wqkv = (const float*)d_in[1];   // (1024,3072)
    const float* wout = (const float*)d_in[2];   // (1024,1024)
    const float* bout = (const float*)d_in[3];   // (1024,)
    float* out = (float*)d_out;
    unsigned short* qkv = (unsigned short*)d_ws; // 16384*3072 bf16 = 96 MB

    qkv_gemm_kernel<<<dim3(N3 / 128, N_TOK / 128), 256, 0, stream>>>(x, wqkv, qkv);
    attn_kernel<<<dim3(HEADS, N_TOK / WS_), 256, 0, stream>>>(qkv);
    out_gemm_kernel<<<dim3(DIM / 128, N_TOK / 128), 256, 0, stream>>>(qkv, wout, bout, out);
}
